// GraphAttentionLayer111111_2791728743043
// MI455X (gfx1250) — matrix operand order repaired and checked by reading, not compiled, not run on hardware
//
#include <hip/hip_runtime.h>
#include <hip/hip_bf16.h>

// ---------------------------------------------------------------------------
// GAT layer, fused flash-attention style for MI455X (gfx1250, wave32, WMMA)
//   Phase A: h = inp@W  (v_wmma_f32_16x16x4_f32), f1 = h.a1, f2 = h.a2,
//            hbT = bf16 transpose of h  (phase-B B-operand = two 16B loads per lane)
//   Phase B: split-j online-softmax attention, 4 waves/block each owning a
//            512-wide j segment (v_wmma_f32_16x16x32_bf16), LDS combine.
// ---------------------------------------------------------------------------

#define IN_F  128
#define OUT_F 64
#define NN    2048
#define BB    8
#define ALPHA 0.2f
#define NEGINF -1e12f
#define NWAVE 4              // j-split factor (waves per phase-B block)
#define JSEG  (NN / NWAVE)   // 512 j's per wave

typedef __attribute__((ext_vector_type(2)))  float  v2f;
typedef __attribute__((ext_vector_type(8)))  float  v8f;
typedef __attribute__((ext_vector_type(16))) __bf16 v16bf;

__device__ __forceinline__ unsigned f2bf_rne(float x) {
    unsigned u = __builtin_bit_cast(unsigned, x);
    unsigned r = u + 0x7FFFu + ((u >> 16) & 1u);  // round-to-nearest-even
    return r >> 16;
}

// ===========================================================================
// Phase A: one wave computes a 16-row tile of h (16 x 64), plus f1/f2.
//   A layout (f32 16x4):  lanes 0-15: VGPR0=K0,VGPR1=K1 ; lanes 16-31: K2,K3
//   C layout:             VGPR r, lane L -> row = r + 8*(L/16), col = L%16
// ===========================================================================
__global__ __launch_bounds__(32)
void gat_phaseA(const float* __restrict__ inp, const float* __restrict__ W,
                const float* __restrict__ avec,
                unsigned short* __restrict__ hbT,
                float* __restrict__ f1, float* __restrict__ f2)
{
    const int b  = blockIdx.y;
    const int i0 = blockIdx.x * 16;
    const int ln = threadIdx.x;      // 0..31
    const int g  = ln >> 4;          // lane half
    const int li = ln & 15;

    const float* inpRow = inp + ((size_t)(b * NN + i0 + li)) * IN_F;

    v8f acc[4] = {v8f{0}, v8f{0}, v8f{0}, v8f{0}};

    for (int k = 0; k < IN_F; k += 4) {
        const int kk = k + 2 * g;
        v2f a;
        a.x = inpRow[kk];
        a.y = inpRow[kk + 1];
#pragma unroll
        for (int ft = 0; ft < 4; ++ft) {
            const int col = ft * 16 + li;
            v2f bv;
            bv.x = W[kk * OUT_F + col];
            bv.y = W[(kk + 1) * OUT_F + col];
            acc[ft] = __builtin_amdgcn_wmma_f32_16x16x4_f32(
                false, a, false, bv, (short)0, acc[ft], false, false);
        }
    }

    // --- f1/f2: per-row dot with a1/a2, reduce across the 16 lanes of a half
    float a1c[4], a2c[4];
#pragma unroll
    for (int ft = 0; ft < 4; ++ft) {
        a1c[ft] = avec[ft * 16 + li];
        a2c[ft] = avec[OUT_F + ft * 16 + li];
    }
#pragma unroll
    for (int r = 0; r < 8; ++r) {
        float p1 = 0.f, p2 = 0.f;
#pragma unroll
        for (int ft = 0; ft < 4; ++ft) {
            const float v = acc[ft][r];
            p1 += v * a1c[ft];
            p2 += v * a2c[ft];
        }
#pragma unroll
        for (int off = 8; off > 0; off >>= 1) {
            p1 += __shfl_xor(p1, off, 32);
            p2 += __shfl_xor(p2, off, 32);
        }
        if (li == 0) {
            const int row = i0 + r + 8 * g;
            f1[b * NN + row] = p1;
            f2[b * NN + row] = p2;
        }
    }

    // --- store bf16 transpose hbT[b][col][i]; r=0..7 are 8 consecutive i's
#pragma unroll
    for (int ft = 0; ft < 4; ++ft) {
        uint4 pk;
        pk.x = f2bf_rne(acc[ft][0]) | (f2bf_rne(acc[ft][1]) << 16);
        pk.y = f2bf_rne(acc[ft][2]) | (f2bf_rne(acc[ft][3]) << 16);
        pk.z = f2bf_rne(acc[ft][4]) | (f2bf_rne(acc[ft][5]) << 16);
        pk.w = f2bf_rne(acc[ft][6]) | (f2bf_rne(acc[ft][7]) << 16);
        const int col = ft * 16 + li;
        const size_t idx = ((size_t)(b * OUT_F + col)) * NN + i0 + 8 * g;
        *reinterpret_cast<uint4*>(hbT + idx) = pk;   // 16B aligned
    }
}

// ===========================================================================
// Phase B: block = 4 waves; wave w owns j in [w*512, (w+1)*512).
//   bf16 A layout (16x32): lane L, half g=L/16 owns K in {8g..8g+7} and
//                          {16+8g..16+8g+7}; v16bf elems in that order.
//   bf16 B layout (32x16): same K order as A (half g: K 8g..8g+7, 16+8g..16+8g+7).
//   Partial (m,l,acc) states combined exactly via LDS rescale-and-sum.
// ===========================================================================
__global__ __launch_bounds__(32 * NWAVE)
void gat_phaseB(const int* __restrict__ adj,
                const unsigned short* __restrict__ hbT,
                const float* __restrict__ f1, const float* __restrict__ f2,
                float* __restrict__ out)
{
    __shared__ float sm_m[NWAVE * 16];
    __shared__ float sm_l[NWAVE * 16];
    __shared__ float sm_acc[NWAVE * 4 * 8 * 32];   // [w][ft][r][lane], 16 KB

    const int b  = blockIdx.y;
    const int i0 = blockIdx.x * 16;
    const int wv = threadIdx.x >> 5;   // wave id 0..3
    const int ln = threadIdx.x & 31;
    const int g  = ln >> 4;
    const int li = ln & 15;

    const int   row  = i0 + li;              // softmax-stats row of this lane
    const float f1r  = f1[b * NN + row];
    const int*  adjR = adj + (size_t)row * NN;
    const float* f2b = f2 + b * NN;

    float m = -3.0e38f;                       // running max (finite sentinel)
    float l = 0.f;                            // running sum
    v8f acc[4] = {v8f{0}, v8f{0}, v8f{0}, v8f{0}};

    const int jbeg = wv * JSEG;
    const int jend = jbeg + JSEG;

    for (int j0 = jbeg; j0 < jend; j0 += 32) {
        // ---- masked leaky-relu logits for this lane's 16 K slots ----
        float e[16];
#pragma unroll
        for (int half = 0; half < 2; ++half) {
            const int base = j0 + 16 * half + 8 * g;
            const float4 fa = *reinterpret_cast<const float4*>(&f2b[base]);
            const float4 fb = *reinterpret_cast<const float4*>(&f2b[base + 4]);
            const int4   m0 = *reinterpret_cast<const int4*>(&adjR[base]);
            const int4   m1 = *reinterpret_cast<const int4*>(&adjR[base + 4]);
            const float tv[8] = {fa.x, fa.y, fa.z, fa.w, fb.x, fb.y, fb.z, fb.w};
            const int   mv[8] = {m0.x, m0.y, m0.z, m0.w, m1.x, m1.y, m1.z, m1.w};
#pragma unroll
            for (int q = 0; q < 8; ++q) {
                float ee = f1r + tv[q];
                ee = ee > 0.f ? ee : ALPHA * ee;
                e[half * 8 + q] = (mv[q] > 0) ? ee : NEGINF;
            }
        }

        // ---- online softmax update ----
        float tmax = e[0];
#pragma unroll
        for (int q = 1; q < 16; ++q) tmax = fmaxf(tmax, e[q]);
        tmax = fmaxf(tmax, __shfl_xor(tmax, 16, 32));     // combine lane pair
        const float mnew  = fmaxf(m, tmax);
        const float scale = __expf(m - mnew);
        m = mnew;

        float ssum = 0.f;
        unsigned short ps[16];
#pragma unroll
        for (int q = 0; q < 16; ++q) {
            const float p = __expf(e[q] - m);
            ssum += p;
            ps[q] = (unsigned short)f2bf_rne(p);
        }
        const float stot = ssum + __shfl_xor(ssum, 16, 32);
        l = l * scale + stot;

        // rescale accumulator rows (acc row of VGPR r = r + 8g)
#pragma unroll
        for (int r = 0; r < 8; ++r) {
            const float fac = __shfl(scale, r + 8 * g, 32);
#pragma unroll
            for (int ft = 0; ft < 4; ++ft) acc[ft][r] *= fac;
        }

        // ---- pack P (A operand) and run 4 bf16 WMMAs over F_out slabs ----
        union { unsigned u[8]; v16bf v; } pa;
#pragma unroll
        for (int q = 0; q < 8; ++q)
            pa.u[q] = (unsigned)ps[2 * q] | ((unsigned)ps[2 * q + 1] << 16);

#pragma unroll
        for (int ft = 0; ft < 4; ++ft) {
            const int col = ft * 16 + li;
            const size_t bidx = ((size_t)(b * OUT_F + col)) * NN + j0 + 8 * g;
            union { uint4 u4[2]; v16bf v; } bv;
            bv.u4[0] = *reinterpret_cast<const uint4*>(&hbT[bidx]);
            bv.u4[1] = *reinterpret_cast<const uint4*>(&hbT[bidx + 16]);
            acc[ft] = __builtin_amdgcn_wmma_f32_16x16x32_bf16(
                false, pa.v, false, bv.v, (short)0, acc[ft], false, false);
        }
    }

    // ---- combine the NWAVE partial softmax states via LDS ----
    if (ln < 16) {                     // stats replicated in both lane halves
        sm_m[wv * 16 + ln] = m;
        sm_l[wv * 16 + ln] = l;
    }
    __syncthreads();

    // global max + rescaled denominator for this lane's row
    float mtot = -3.0e38f;
#pragma unroll
    for (int w = 0; w < NWAVE; ++w) mtot = fmaxf(mtot, sm_m[w * 16 + li]);
    float ltot = 0.f;
#pragma unroll
    for (int w = 0; w < NWAVE; ++w)
        ltot += sm_l[w * 16 + li] * __expf(sm_m[w * 16 + li] - mtot);
    const float sc = __expf(m - mtot);   // this wave's correction factor

    // scale own accumulator and stage it in LDS
#pragma unroll
    for (int r = 0; r < 8; ++r) {
        const float fac = __shfl(sc, r + 8 * g, 32);
#pragma unroll
        for (int ft = 0; ft < 4; ++ft)
            sm_acc[(((wv * 4 + ft) * 8) + r) * 32 + ln] = acc[ft][r] * fac;
    }
    __syncthreads();

    // wave wv reduces + writes output slab ft == wv (cols 16*wv .. 16*wv+15)
    const int ft = wv;
#pragma unroll
    for (int r = 0; r < 8; ++r) {
        const float lrow = __shfl(ltot, r + 8 * g, 32);
        float s = 0.f;
#pragma unroll
        for (int w = 0; w < NWAVE; ++w)
            s += sm_acc[(((w * 4 + ft) * 8) + r) * 32 + ln];
        float v = s / lrow;
        v = v > 0.f ? v : 0.f;
        const int orow = i0 + r + 8 * g;
        out[((size_t)(b * NN + orow)) * OUT_F + ft * 16 + li] = v;
    }
}

// ===========================================================================
extern "C" void kernel_launch(void* const* d_in, const int* in_sizes, int n_in,
                              void* d_out, int out_size, void* d_ws, size_t ws_size,
                              hipStream_t stream) {
    const float* inp  = (const float*)d_in[0];   // [B,N,128] f32
    const int*   adj  = (const int*)  d_in[1];   // [N,N] i32
    const float* W    = (const float*)d_in[2];   // [128,64] f32
    const float* avec = (const float*)d_in[3];   // [128,1] f32
    float*       out  = (float*)d_out;           // [B,N,64] f32

    // workspace: hbT (bf16, B*64*N) | f1 (f32, B*N) | f2 (f32, B*N)
    unsigned short* hbT = (unsigned short*)d_ws;
    float* f1 = (float*)((char*)d_ws + (size_t)BB * OUT_F * NN * sizeof(unsigned short));
    float* f2 = f1 + (size_t)BB * NN;

    dim3 grid(NN / 16, BB);
    gat_phaseA<<<grid, 32, 0, stream>>>(inp, W, avec, hbT, f1, f2);
    gat_phaseB<<<grid, 32 * NWAVE, 0, stream>>>(adj, hbT, f1, f2, out);
}
